// GraphTransformerQAGNN_69930657513934
// MI455X (gfx1250) — matrix operand order repaired and checked by reading, not compiled, not run on hardware
//
#include <hip/hip_runtime.h>
#include <hip/hip_bf16.h>

#define N_NODES 16384
#define N_EDGES 65536
#define N_GRAPHS 64
#define HID 768
#define DD 256
#define HEADS 4
#define HD 1024   /* HEADS*DD */

typedef __bf16 v16bf __attribute__((ext_vector_type(16)));
typedef float  v8f   __attribute__((ext_vector_type(8)));
typedef unsigned int u32x4 __attribute__((ext_vector_type(4)));

union FragBF { u32x4 u[2]; v16bf v; };

__device__ __forceinline__ unsigned short f2bf(float f) {
  unsigned int u = __float_as_uint(f);
  u += 0x7FFFu + ((u >> 16) & 1u);          // round-to-nearest-even
  return (unsigned short)(u >> 16);
}

__device__ __forceinline__ void atomicMaxF(float* addr, float val) {
  int* ia = (int*)addr;
  int old = __float_as_int(*addr);
  while (__int_as_float(old) < val) {
    int assumed = old;
    old = atomicCAS(ia, assumed, __float_as_int(val));
    if (old == assumed) break;
  }
}

// ---------------------------------------------------------------- utilities
__global__ void fill_f32(float* p, float v, int n) {
  int i = blockIdx.x * blockDim.x + threadIdx.x;
  if (i < n) p[i] = v;
}

// src fp32 [R][C] row-major -> dst bf16 [C][R] row-major (i.e. transposed)
__global__ void transpose_to_bf16(const float* __restrict__ src,
                                  unsigned short* __restrict__ dst, int R, int C) {
  int i = blockIdx.x * blockDim.x + threadIdx.x;
  if (i >= R * C) return;
  int r = i / C, c = i % C;
  dst[(size_t)c * R + r] = f2bf(src[i]);
}

// ---------------------------------------------------------------- WMMA GEMM
// C[M,N] = A_bf16[M,K] @ Bt_bf16[N,K]^T + bias[N]; optional bf16 mirror of C.
// Wave tile 16x64 (4 accumulators), block = 8 waves -> 128x64 block tile.
// Requires M%128==0, N%64==0, K%32==0 (true for all call sites here).
__global__ void __launch_bounds__(256) gemm_bf16_wmma(
    const unsigned short* __restrict__ A,
    const unsigned short* __restrict__ Bt,
    const float* __restrict__ bias,
    float* __restrict__ C,
    unsigned short* __restrict__ Cbf,
    int M, int N, int K) {
  const int wave  = threadIdx.x >> 5;
  const int lane  = threadIdx.x & 31;
  const int mBase = (blockIdx.y * 8 + wave) * 16;
  const int nBase = blockIdx.x * 64;
  const int l15   = lane & 15;
  const int hi    = lane >> 4;            // 0 or 1
  const int koffA = hi * 16;              // A: K chunk  {koff..+15}
  const int koffB = hi * 16;              // B: K chunk  {koff..+15}
  const size_t rowA  = (size_t)(mBase + l15) * K;
  const size_t rowB0 = (size_t)(nBase +  0 + l15) * K;
  const size_t rowB1 = (size_t)(nBase + 16 + l15) * K;
  const size_t rowB2 = (size_t)(nBase + 32 + l15) * K;
  const size_t rowB3 = (size_t)(nBase + 48 + l15) * K;

  v8f acc0 = {}, acc1 = {}, acc2 = {}, acc3 = {};
  for (int kk = 0; kk < K; kk += 32) {
    FragBF a, b0, b1, b2, b3;
    const u32x4* pa = (const u32x4*)(A + rowA + kk + koffA);
    a.u[0] = pa[0]; a.u[1] = pa[1];                 // +0 and +8 elements
    const u32x4* pb0 = (const u32x4*)(Bt + rowB0 + kk + koffB);
    b0.u[0] = pb0[0]; b0.u[1] = pb0[1];
    const u32x4* pb1 = (const u32x4*)(Bt + rowB1 + kk + koffB);
    b1.u[0] = pb1[0]; b1.u[1] = pb1[1];
    const u32x4* pb2 = (const u32x4*)(Bt + rowB2 + kk + koffB);
    b2.u[0] = pb2[0]; b2.u[1] = pb2[1];
    const u32x4* pb3 = (const u32x4*)(Bt + rowB3 + kk + koffB);
    b3.u[0] = pb3[0]; b3.u[1] = pb3[1];
    acc0 = __builtin_amdgcn_wmma_f32_16x16x32_bf16(false, a.v, false, b0.v, (short)0, acc0, false, false);
    acc1 = __builtin_amdgcn_wmma_f32_16x16x32_bf16(false, a.v, false, b1.v, (short)0, acc1, false, false);
    acc2 = __builtin_amdgcn_wmma_f32_16x16x32_bf16(false, a.v, false, b2.v, (short)0, acc2, false, false);
    acc3 = __builtin_amdgcn_wmma_f32_16x16x32_bf16(false, a.v, false, b3.v, (short)0, acc3, false, false);
  }
  // epilogue: C/D layout row = r + 8*hi, col = nBase + 16*t + l15
  #pragma unroll
  for (int r = 0; r < 8; ++r) {
    int row = mBase + r + hi * 8;
    size_t base = (size_t)row * N + nBase + l15;
    float v0 = acc0[r] + bias[nBase +  0 + l15];
    float v1 = acc1[r] + bias[nBase + 16 + l15];
    float v2 = acc2[r] + bias[nBase + 32 + l15];
    float v3 = acc3[r] + bias[nBase + 48 + l15];
    C[base +  0] = v0;  C[base + 16] = v1;
    C[base + 32] = v2;  C[base + 48] = v3;
    if (Cbf) {
      Cbf[base +  0] = f2bf(v0);  Cbf[base + 16] = f2bf(v1);
      Cbf[base + 32] = f2bf(v2);  Cbf[base + 48] = f2bf(v3);
    }
  }
}

// ---------------------------------------------------------------- preprocessing
__global__ void text_norms(const float* __restrict__ text, float* __restrict__ tnorm) {
  int wid = (blockIdx.x * blockDim.x + threadIdx.x) >> 5;
  int lane = threadIdx.x & 31;
  if (wid >= N_GRAPHS) return;
  float s = 0.f;
  for (int j = lane; j < HID; j += 32) { float v = text[(size_t)wid * HID + j]; s += v * v; }
  for (int o = 16; o; o >>= 1) s += __shfl_xor(s, o, 32);
  if (lane == 0) tnorm[wid] = sqrtf(s);
}

// cosine-relevance reweight, emit bf16 node features [N,768]
__global__ void relevance_bf16(const float* __restrict__ text, const float* __restrict__ nodex,
                               const int* __restrict__ batch, const float* __restrict__ tnorm,
                               unsigned short* __restrict__ nf) {
  int wid = (blockIdx.x * blockDim.x + threadIdx.x) >> 5;
  int lane = threadIdx.x & 31;
  if (wid >= N_NODES) return;
  int b = batch[wid];
  const float* tx = text + (size_t)b * HID;
  const float* nx = nodex + (size_t)wid * HID;
  float num = 0.f, nn = 0.f;
  for (int j = lane; j < HID; j += 32) { float a = tx[j], c = nx[j]; num += a * c; nn += c * c; }
  for (int o = 16; o; o >>= 1) { num += __shfl_xor(num, o, 32); nn += __shfl_xor(nn, o, 32); }
  float rel = num / fmaxf(tnorm[b] * sqrtf(nn), 1e-8f);
  for (int j = lane; j < HID; j += 32) nf[(size_t)wid * HID + j] = f2bf(nx[j] * rel);
}

// tproj_g[g,:] = text_cls[g,:] @ node_proj_w + b   (fp32, tiny)
__global__ void text_proj(const float* __restrict__ text, const float* __restrict__ W,
                          const float* __restrict__ bias, float* __restrict__ out) {
  int i = blockIdx.x * blockDim.x + threadIdx.x;
  int g = i >> 8, o = i & 255;
  float acc = bias[o];
  const float* tx = text + (size_t)g * HID;
  for (int j = 0; j < HID; ++j) acc += tx[j] * W[(size_t)j * DD + o];
  out[i] = acc;
}

// ---------------------------------------------------------------- edge attention
__global__ void edge_scores(const float* __restrict__ q, const float* __restrict__ k,
                            const int* __restrict__ src, const int* __restrict__ dst,
                            float* __restrict__ score, float* __restrict__ mmax) {
  int wid = (blockIdx.x * blockDim.x + threadIdx.x) >> 5;
  int lane = threadIdx.x & 31;
  if (wid >= N_EDGES) return;
  int s = src[wid], d = dst[wid];
  const float4* qp = (const float4*)(q + (size_t)d * HD);
  const float4* kp = (const float4*)(k + (size_t)s * HD);
  #pragma unroll
  for (int hh = 0; hh < HEADS; ++hh) {
    float acc = 0.f;
    for (int j = lane; j < 64; j += 32) {
      float4 a = qp[hh * 64 + j], b = kp[hh * 64 + j];
      acc += a.x * b.x + a.y * b.y + a.z * b.z + a.w * b.w;
    }
    for (int o = 16; o; o >>= 1) acc += __shfl_xor(acc, o, 32);
    if (lane == 0) {
      acc *= 0.0625f;                     // 1/sqrt(256)
      score[wid * 4 + hh] = acc;
      atomicMaxF(&mmax[d * 4 + hh], acc);
    }
  }
}

__global__ void edge_exp(float* __restrict__ score, const float* __restrict__ mmax,
                         const int* __restrict__ dst, float* __restrict__ den) {
  int i = blockIdx.x * blockDim.x + threadIdx.x;
  if (i >= N_EDGES * HEADS) return;
  int e = i >> 2, hh = i & 3;
  int d = dst[e];
  float ex = expf(score[i] - mmax[d * 4 + hh]);
  score[i] = ex;
  atomicAdd(&den[d * 4 + hh], ex);
}

__global__ void edge_agg(const float* __restrict__ score, const float* __restrict__ v,
                         const int* __restrict__ src, const int* __restrict__ dst,
                         float* __restrict__ num) {
  int e = blockIdx.x;
  int t = threadIdx.x;                    // 256 threads x 4 floats = 1024
  int s = src[e], d = dst[e];
  int base = t * 4;
  float ex = score[e * 4 + (base >> 8)];
  float4 vv = *(const float4*)(v + (size_t)s * HD + base);
  float* np = num + (size_t)d * HD + base;
  atomicAdd(np + 0, ex * vv.x);
  atomicAdd(np + 1, ex * vv.y);
  atomicAdd(np + 2, ex * vv.z);
  atomicAdd(np + 3, ex * vv.w);
}

__global__ void combine_heads(const float* __restrict__ num, const float* __restrict__ den,
                              const float* __restrict__ sk, float* __restrict__ h) {
  int i = blockIdx.x * blockDim.x + threadIdx.x;   // N*256 exact
  int n = i >> 8, c = i & 255;
  float acc = 0.f;
  #pragma unroll
  for (int hh = 0; hh < HEADS; ++hh)
    acc += num[(size_t)n * HD + hh * DD + c] / (den[n * 4 + hh] + 1e-16f);
  h[i] = 0.25f * acc + sk[i];
}

// layernorm (+optional residual) + relu; writes fp32 x and bf16 mirror
__global__ void ln_relu(const float* __restrict__ h, const float* __restrict__ res,
                        const float* __restrict__ g, const float* __restrict__ be,
                        float* __restrict__ xo, unsigned short* __restrict__ xbf) {
  int wid = (blockIdx.x * blockDim.x + threadIdx.x) >> 5;
  int lane = threadIdx.x & 31;
  if (wid >= N_NODES) return;
  size_t base = (size_t)wid * DD;
  float vals[8]; float s = 0.f;
  #pragma unroll
  for (int j = 0; j < 8; ++j) {
    int c = lane + 32 * j;
    float t = h[base + c];
    if (res) t += res[base + c];
    vals[j] = t; s += t;
  }
  for (int o = 16; o; o >>= 1) s += __shfl_xor(s, o, 32);
  float mu = s * (1.0f / 256.0f);
  float q = 0.f;
  #pragma unroll
  for (int j = 0; j < 8; ++j) { float d = vals[j] - mu; q += d * d; }
  for (int o = 16; o; o >>= 1) q += __shfl_xor(q, o, 32);
  float inv = rsqrtf(q * (1.0f / 256.0f) + 1e-5f);
  #pragma unroll
  for (int j = 0; j < 8; ++j) {
    int c = lane + 32 * j;
    float y = fmaxf((vals[j] - mu) * inv * g[c] + be[c], 0.f);
    xo[base + c] = y;
    xbf[base + c] = f2bf(y);
  }
}

// ---------------------------------------------------------------- pooling + head
__global__ void sem_scores(const float* __restrict__ x, const float* __restrict__ tproj,
                           const int* __restrict__ batch, float* __restrict__ exn,
                           float* __restrict__ dsem, float* __restrict__ cnt) {
  int wid = (blockIdx.x * blockDim.x + threadIdx.x) >> 5;
  int lane = threadIdx.x & 31;
  if (wid >= N_NODES) return;
  int b = batch[wid];
  float s = 0.f;
  for (int c = lane; c < DD; c += 32) s += x[(size_t)wid * DD + c] * tproj[b * DD + c];
  for (int o = 16; o; o >>= 1) s += __shfl_xor(s, o, 32);
  if (lane == 0) {
    float e = expf(s);
    exn[wid] = e;
    atomicAdd(&dsem[b], e);
    atomicAdd(&cnt[b], 1.0f);
  }
}

__global__ void pool_accum(const float* __restrict__ x, const float* __restrict__ exn,
                           const int* __restrict__ batch, float* __restrict__ ps,
                           float* __restrict__ psem) {
  int i = blockIdx.x * blockDim.x + threadIdx.x;   // N*256 exact
  int n = i >> 8, c = i & 255;
  int b = batch[n];
  float v = x[i];
  atomicAdd(&ps[b * DD + c], v);
  atomicAdd(&psem[b * DD + c], v * exn[n]);
}

__global__ void classifier(const float* __restrict__ ps, const float* __restrict__ psem,
                           const float* __restrict__ cnt, const float* __restrict__ dsem,
                           const float* __restrict__ text, const float* __restrict__ w,
                           const float* __restrict__ b, float* __restrict__ out) {
  int g = blockIdx.x, t = threadIdx.x;
  float invc = 1.0f / fmaxf(cnt[g], 1.0f);
  float invd = 1.0f / (dsem[g] + 1e-8f);
  float acc = 0.f;
  for (int i = t; i < 2 * DD + HID; i += 256) {
    float c;
    if (i < DD)          c = ps[g * DD + i] * invc;
    else if (i < 2 * DD) c = psem[g * DD + (i - DD)] * invd;
    else                 c = text[(size_t)g * HID + (i - 2 * DD)];
    acc += c * w[i];
  }
  __shared__ float red[256];
  red[t] = acc;
  __syncthreads();
  for (int s = 128; s > 0; s >>= 1) { if (t < s) red[t] += red[t + s]; __syncthreads(); }
  if (t == 0) out[g] = red[0] + b[0];
}

// ---------------------------------------------------------------- launcher
extern "C" void kernel_launch(void* const* d_in, const int* in_sizes, int n_in,
                              void* d_out, int out_size, void* d_ws, size_t ws_size,
                              hipStream_t stream) {
  (void)in_sizes; (void)n_in; (void)out_size; (void)ws_size;
  const float* text = (const float*)d_in[0];
  const float* nodex= (const float*)d_in[1];
  const float* Wp   = (const float*)d_in[2];
  const float* bp   = (const float*)d_in[3];
  const float* Wq   = (const float*)d_in[4];
  const float* bq   = (const float*)d_in[5];
  const float* Wk   = (const float*)d_in[6];
  const float* bk   = (const float*)d_in[7];
  const float* Wv   = (const float*)d_in[8];
  const float* bv   = (const float*)d_in[9];
  const float* Wsk  = (const float*)d_in[10];
  const float* bsk  = (const float*)d_in[11];
  const float* gam  = (const float*)d_in[12];
  const float* bet  = (const float*)d_in[13];
  const float* clfw = (const float*)d_in[14];
  const float* clfb = (const float*)d_in[15];
  const int*   eidx = (const int*)d_in[16];
  const int*   batch= (const int*)d_in[17];
  const int* src = eidx;
  const int* dst = eidx + N_EDGES;

  char* ws = (char*)d_ws;
  size_t off = 0;
  auto alloc = [&](size_t bytes) -> char* {
    char* p = ws + off;
    off = (off + bytes + 255) & ~(size_t)255;
    return p;
  };
  unsigned short* nf  = (unsigned short*)alloc((size_t)N_NODES * HID * 2);
  unsigned short* WpT = (unsigned short*)alloc((size_t)DD * HID * 2);
  unsigned short *WqT[2], *WkT[2], *WvT[2], *WsT[2];
  for (int l = 0; l < 2; ++l) {
    WqT[l] = (unsigned short*)alloc((size_t)HD * DD * 2);
    WkT[l] = (unsigned short*)alloc((size_t)HD * DD * 2);
    WvT[l] = (unsigned short*)alloc((size_t)HD * DD * 2);
    WsT[l] = (unsigned short*)alloc((size_t)DD * DD * 2);
  }
  float* xf   = (float*)alloc((size_t)N_NODES * DD * 4);
  unsigned short* xbf = (unsigned short*)alloc((size_t)N_NODES * DD * 2);
  float* qb   = (float*)alloc((size_t)N_NODES * HD * 4);
  float* kb   = (float*)alloc((size_t)N_NODES * HD * 4);
  float* vb   = (float*)alloc((size_t)N_NODES * HD * 4);
  float* skb  = (float*)alloc((size_t)N_NODES * DD * 4);
  float* hb   = (float*)alloc((size_t)N_NODES * DD * 4);
  float* scoreb = (float*)alloc((size_t)N_EDGES * HEADS * 4);
  float* mb   = (float*)alloc((size_t)N_NODES * HEADS * 4);
  float* denb = (float*)alloc((size_t)N_NODES * HEADS * 4);
  float* tnorm  = (float*)alloc(N_GRAPHS * 4);
  float* tprojb = (float*)alloc((size_t)N_GRAPHS * DD * 4);
  float* exn  = (float*)alloc((size_t)N_NODES * 4);
  const int poolN = 2 * N_GRAPHS * DD + 2 * N_GRAPHS;
  float* poolreg = (float*)alloc((size_t)poolN * 4);
  float* ps   = poolreg;
  float* psem = poolreg + N_GRAPHS * DD;
  float* cnt  = poolreg + 2 * N_GRAPHS * DD;
  float* dsem = cnt + N_GRAPHS;
  float* numb = qb;   // q no longer needed once scores exist -> reuse as aggregate

  // --- preprocessing
  text_norms<<<(N_GRAPHS * 32) / 256, 256, 0, stream>>>(text, tnorm);
  relevance_bf16<<<(N_NODES * 32) / 256, 256, 0, stream>>>(text, nodex, batch, tnorm, nf);
  transpose_to_bf16<<<(HID * DD) / 256, 256, 0, stream>>>(Wp, WpT, HID, DD);
  for (int l = 0; l < 2; ++l) {
    transpose_to_bf16<<<(DD * HD) / 256, 256, 0, stream>>>(Wq + (size_t)l * DD * HD, WqT[l], DD, HD);
    transpose_to_bf16<<<(DD * HD) / 256, 256, 0, stream>>>(Wk + (size_t)l * DD * HD, WkT[l], DD, HD);
    transpose_to_bf16<<<(DD * HD) / 256, 256, 0, stream>>>(Wv + (size_t)l * DD * HD, WvT[l], DD, HD);
    transpose_to_bf16<<<(DD * DD) / 256, 256, 0, stream>>>(Wsk + (size_t)l * DD * DD, WsT[l], DD, DD);
  }
  // x = nf @ Wp + bp
  gemm_bf16_wmma<<<dim3(DD / 64, N_NODES / 128), 256, 0, stream>>>(
      nf, WpT, bp, xf, xbf, N_NODES, DD, HID);
  text_proj<<<(N_GRAPHS * DD) / 256, 256, 0, stream>>>(text, Wp, bp, tprojb);

  // --- two TransformerConv layers
  for (int l = 0; l < 2; ++l) {
    gemm_bf16_wmma<<<dim3(HD / 64, N_NODES / 128), 256, 0, stream>>>(
        xbf, WqT[l], bq + l * HD, qb, nullptr, N_NODES, HD, DD);
    gemm_bf16_wmma<<<dim3(HD / 64, N_NODES / 128), 256, 0, stream>>>(
        xbf, WkT[l], bk + l * HD, kb, nullptr, N_NODES, HD, DD);
    gemm_bf16_wmma<<<dim3(HD / 64, N_NODES / 128), 256, 0, stream>>>(
        xbf, WvT[l], bv + l * HD, vb, nullptr, N_NODES, HD, DD);
    gemm_bf16_wmma<<<dim3(DD / 64, N_NODES / 128), 256, 0, stream>>>(
        xbf, WsT[l], bsk + l * DD, skb, nullptr, N_NODES, DD, DD);

    fill_f32<<<(N_NODES * HEADS) / 256, 256, 0, stream>>>(mb, -3.0e38f, N_NODES * HEADS);
    fill_f32<<<(N_NODES * HEADS) / 256, 256, 0, stream>>>(denb, 0.f, N_NODES * HEADS);
    edge_scores<<<(N_EDGES * 32) / 256, 256, 0, stream>>>(qb, kb, src, dst, scoreb, mb);
    edge_exp<<<(N_EDGES * HEADS) / 256, 256, 0, stream>>>(scoreb, mb, dst, denb);
    fill_f32<<<(N_NODES * HD) / 256, 256, 0, stream>>>(numb, 0.f, N_NODES * HD);
    edge_agg<<<N_EDGES, 256, 0, stream>>>(scoreb, vb, src, dst, numb);
    combine_heads<<<(N_NODES * DD) / 256, 256, 0, stream>>>(numb, denb, skb, hb);
    ln_relu<<<(N_NODES * 32) / 256, 256, 0, stream>>>(
        hb, l == 0 ? xf : nullptr, gam + l * DD, bet + l * DD, xf, xbf);
  }

  // --- pooling + classifier
  fill_f32<<<(poolN + 255) / 256, 256, 0, stream>>>(poolreg, 0.f, poolN);
  sem_scores<<<(N_NODES * 32) / 256, 256, 0, stream>>>(xf, tprojb, batch, exn, dsem, cnt);
  pool_accum<<<(N_NODES * DD) / 256, 256, 0, stream>>>(xf, exn, batch, ps, psem);
  classifier<<<N_GRAPHS, 256, 0, stream>>>(ps, psem, cnt, dsem, text, clfw, clfb, (float*)d_out);
}
